// NMT_39951785787705
// MI455X (gfx1250) — compile-verified, matrix operand order checked by reading, not run on hardware
//
#include <hip/hip_runtime.h>
#include <hip/hip_bf16.h>

typedef float v2f __attribute__((ext_vector_type(2)));
typedef float v8f __attribute__((ext_vector_type(8)));
typedef __bf16 v16bf __attribute__((ext_vector_type(16)));
typedef __bf16 v8bf __attribute__((ext_vector_type(8)));

#define S_ 96
#define T_ 96
#define B_ 8
#define V_ 32000
#define E_ 256
#define H_ 512
#define G4_ 2048
#define NSTEP_ 192
#define NROW_ 1536   // 192*8 rows of H_all / Xg
#define NDEC_ 768    // 96*8 decoder rows
#define NEGV (-1e30f)

__device__ __forceinline__ v2f ld2(const float* p) { return *(const v2f*)p; }

__device__ __forceinline__ v8f wmma_f32(v2f a, v2f b, v8f c) {
  // D = A(16x4,f32) * B(4x16,f32) + C(16x16,f32)
  return __builtin_amdgcn_wmma_f32_16x16x4_f32(false, a, false, b, (short)0, c,
                                               false, false);
}

// bf16 16x32 fragment: lane's 16 values = two contiguous 8-elem K-runs
// (ISA 7.12.2): [k0 + 8*half, +8) and [k0 + 16 + 8*half, +8).
union FragBF {
  v16bf v;
  v8bf h[2];
};
__device__ __forceinline__ v16bf ldfrag_bf(const __bf16* row, int k0,
                                           int half8) {
  FragBF f;
  f.h[0] = *(const v8bf*)(row + k0 + half8);
  f.h[1] = *(const v8bf*)(row + k0 + 16 + half8);
  return f.v;
}

// ---------------------------------------------------------------------------
// K0: zero the accumulators (harness poisons d_out / d_ws with 0xAA)
// ---------------------------------------------------------------------------
__global__ __launch_bounds__(256) void init_kernel(float* out, float* sumexp) {
  int i = blockIdx.x * 256 + threadIdx.x;
  if (i < 16) out[i] = 0.0f;
  if (i < NDEC_) sumexp[i] = 0.0f;
}

// ---------------------------------------------------------------------------
// K1: fused embedding lookup + input-gate GEMM:  Xg = emb_eff[tokens] @ W_ih^T
//     M = 1536 (96 tiles), N = 2048 (128 tiles), K = 256.  (f32 WMMA)
// ---------------------------------------------------------------------------
__global__ __launch_bounds__(256) void xgates_kernel(
    const int* __restrict__ src, const int* __restrict__ tgt,
    const float* __restrict__ emb, const float* __restrict__ W_ih,
    float* __restrict__ Xg) {
  __shared__ float A[16][E_];
  const int m0 = blockIdx.x * 16;
  const int tid = threadIdx.x;
  for (int idx = tid; idx < 16 * E_; idx += 256) {
    int i = idx >> 8;          // E_ == 256
    int k = idx & 255;
    int row = m0 + i;          // row = t*8 + b
    int t = row >> 3, b = row & 7;
    int tok = (t < S_) ? src[t * B_ + b] : tgt[(t - S_) * B_ + b];
    A[i][k] = (tok == 0) ? 0.0f : emb[(size_t)tok * E_ + k];
  }
  __syncthreads();
  const int lane = tid & 31, wave = tid >> 5;
  const int l15 = lane & 15, kh = (lane >> 4) << 1;
  const int rbase = m0 + ((lane >> 4) << 3);
  for (int ti = 0; ti < 2; ++ti) {
    int n0 = (blockIdx.y * 16 + wave * 2 + ti) * 16;
    const float* Brow = W_ih + (size_t)(n0 + l15) * E_;
    v8f acc = {};
    for (int k0 = 0; k0 < E_; k0 += 4) {
      v2f a = ld2(&A[l15][k0 + kh]);
      v2f b = ld2(Brow + k0 + kh);
      acc = wmma_f32(a, b, acc);
    }
    int col = n0 + l15;
#pragma unroll
    for (int r = 0; r < 8; ++r) Xg[(size_t)(rbase + r) * G4_ + col] = acc[r];
  }
}

// ---------------------------------------------------------------------------
// K2: sequential LSTM (recurrent part stays fp32 WMMA: errors would compound
//     through 192 nonlinear steps and this is the latency-bound serial chain).
// ---------------------------------------------------------------------------
__global__ __launch_bounds__(512) void lstm_kernel(
    const float* __restrict__ Xg, const float* __restrict__ W_hh,
    const float* __restrict__ b_ih, const float* __restrict__ b_hh,
    float* __restrict__ H_all) {
  __shared__ float hs[B_][H_];    // 16 KB
  __shared__ float gs[B_][G4_];   // 64 KB
  const int tid = threadIdx.x;
  const int lane = tid & 31, wave = tid >> 5;
  const int l15 = lane & 15, kh = (lane >> 4) << 1;
  const int hi8 = (lane >> 4) << 3;
  float c[B_];
#pragma unroll
  for (int m = 0; m < B_; ++m) c[m] = 0.0f;
  for (int i = tid; i < B_ * H_; i += 512) hs[i >> 9][i & 511] = 0.0f;
  const int j = tid;  // thread owns hidden unit j for all batches
  const float bi0 = b_ih[j] + b_hh[j];
  const float bf0 = b_ih[H_ + j] + b_hh[H_ + j];
  const float bg0 = b_ih[2 * H_ + j] + b_hh[2 * H_ + j];
  const float bo0 = b_ih[3 * H_ + j] + b_hh[3 * H_ + j];
  __syncthreads();
  for (int t = 0; t < NSTEP_; ++t) {
    for (int ti = 0; ti < 8; ++ti) {
      int n0 = (wave * 8 + ti) * 16;
      int col = n0 + l15;
      v8f acc;
#pragma unroll
      for (int r = 0; r < 8; ++r) {
        int row = hi8 + r;
        acc[r] = (row < B_) ? Xg[(size_t)(t * B_ + row) * G4_ + col] : 0.0f;
      }
      const float* Brow = W_hh + (size_t)(n0 + l15) * H_;
      for (int k0 = 0; k0 < H_; k0 += 4) {
        int k = k0 + kh;
        float ax = hs[l15 & 7][k];
        float ay = hs[l15 & 7][k + 1];
        v2f a;
        a.x = (l15 < B_) ? ax : 0.0f;
        a.y = (l15 < B_) ? ay : 0.0f;
        v2f b = ld2(Brow + k);
        acc = wmma_f32(a, b, acc);
      }
      if (lane < 16) {
#pragma unroll
        for (int r = 0; r < 8; ++r) gs[r][col] = acc[r];
      }
    }
    __syncthreads();
#pragma unroll
    for (int m = 0; m < B_; ++m) {
      float gi = gs[m][j] + bi0;
      float gf = gs[m][H_ + j] + bf0;
      float gg = gs[m][2 * H_ + j] + bg0;
      float go = gs[m][3 * H_ + j] + bo0;
      float si = 1.0f / (1.0f + __expf(-gi));
      float sf = 1.0f / (1.0f + __expf(-gf));
      float so = 1.0f / (1.0f + __expf(-go));
      float cv = sf * c[m] + si * tanhf(gg);
      float hv = so * tanhf(cv);
      c[m] = cv;
      hs[m][j] = hv;
      H_all[(size_t)(t * B_ + m) * H_ + j] = hv;
    }
    __syncthreads();
  }
}

// ---------------------------------------------------------------------------
// K2b: precision-downconvert kernels (bf16 fast path for the readout GEMM).
// ---------------------------------------------------------------------------
__global__ __launch_bounds__(256) void wconv_kernel(
    const float* __restrict__ W_read, __bf16* __restrict__ Wbf) {
  const int n = V_ * H_;
  int stride = gridDim.x * 256;
  for (int i = blockIdx.x * 256 + threadIdx.x; i < n; i += stride)
    Wbf[i] = (__bf16)W_read[i];
}

__global__ __launch_bounds__(256) void hconv_kernel(
    const float* __restrict__ H_all, __bf16* __restrict__ Hbf) {
  int i = blockIdx.x * 256 + threadIdx.x;  // grid covers 768*512
  Hbf[i] = (__bf16)H_all[(size_t)NDEC_ * H_ + i];
}

// ---------------------------------------------------------------------------
// K3: Q = tanh(H_dec @ W_att^T).  (f32 WMMA; small, feeds softmax directly)
// ---------------------------------------------------------------------------
__global__ __launch_bounds__(256) void q_kernel(const float* __restrict__ H_all,
                                                const float* __restrict__ W_att,
                                                float* __restrict__ Q) {
  const int tid = threadIdx.x;
  const int lane = tid & 31, wave = tid >> 5;
  const int l15 = lane & 15, kh = (lane >> 4) << 1;
  int tile = blockIdx.x * 8 + wave;   // 0..1535
  int m0 = (tile >> 5) * 16;
  int n0 = (tile & 31) * 16;
  const float* Arow = H_all + (size_t)(NDEC_ + m0 + l15) * H_;
  const float* Brow = W_att + (size_t)(n0 + l15) * H_;
  v8f acc = {};
  for (int k0 = 0; k0 < H_; k0 += 4) {
    v2f a = ld2(Arow + k0 + kh);
    v2f b = ld2(Brow + k0 + kh);
    acc = wmma_f32(a, b, acc);
  }
  int col = n0 + l15;
  int rbase = m0 + ((lane >> 4) << 3);
#pragma unroll
  for (int r = 0; r < 8; ++r)
    Q[(size_t)(rbase + r) * H_ + col] = tanhf(acc[r]);
}

// ---------------------------------------------------------------------------
// K4: per (t,b): attention logits over 193 entries, masked softmax, and the
//     pointer mass at the gold token (sum of weights whose token == gold).
// ---------------------------------------------------------------------------
__global__ __launch_bounds__(256) void att_kernel(
    const float* __restrict__ H_all, const float* __restrict__ Q,
    const float* __restrict__ sent, const int* __restrict__ src,
    const int* __restrict__ tgt, float* __restrict__ a_sent_arr,
    float* __restrict__ pptr_gold) {
  __shared__ float q[H_];
  __shared__ float red[256];
  const int r = blockIdx.x;          // 0..767  (= t*8 + b)
  const int t = r >> 3, b = r & 7;
  const int tid = threadIdx.x;
  for (int i = tid; i < H_; i += 256) q[i] = Q[(size_t)r * H_ + i];
  __syncthreads();
  const int jj = tid;
  float z = NEGV;
  if (jj < 193) {
    const float* v = (jj < 192) ? (H_all + (size_t)(jj * B_ + b) * H_) : sent;
    float acc = 0.0f;
    for (int k = 0; k < H_; ++k) acc += q[k] * v[k];
    bool valid = (jj < S_ + t + 1) || (jj == 192);
    z = valid ? acc : NEGV;
  }
  red[tid] = z;
  __syncthreads();
  for (int s = 128; s > 0; s >>= 1) {
    if (tid < s) red[tid] = fmaxf(red[tid], red[tid + s]);
    __syncthreads();
  }
  float zmax = red[0];
  __syncthreads();
  float e = (jj < 193) ? __expf(z - zmax) : 0.0f;
  red[tid] = e;
  __syncthreads();
  for (int s = 128; s > 0; s >>= 1) {
    if (tid < s) red[tid] += red[tid + s];
    __syncthreads();
  }
  float denom = red[0];
  __syncthreads();
  float a = e / denom;
  int g = tgt[r + 8];                // tgt[(t+1)*8 + b]
  float contrib = 0.0f;
  if (jj < 192) {
    int tok = (jj < S_) ? src[jj * B_ + b] : tgt[(jj - S_) * B_ + b];
    contrib = (tok == g) ? a : 0.0f;
  }
  red[tid] = contrib;
  __syncthreads();
  for (int s = 128; s > 0; s >>= 1) {
    if (tid < s) red[tid] += red[tid + s];
    __syncthreads();
  }
  if (tid == 0) pptr_gold[r] = red[0];
  if (jj == 192) a_sent_arr[r] = a;
}

// ---------------------------------------------------------------------------
// K5: fused readout GEMM + online exp-sum, now on the bf16 matrix pipe:
//     V_WMMA_F32_16X16X32_BF16 (8x the K per issue vs f32 16x16x4; bf16
//     W_read = 32 MB stays L2-resident; half the B bytes per K).
//     A tile (16 rows x 1 KB, contiguous) is staged to LDS with
//     GLOBAL_LOAD_ASYNC_TO_LDS_B128 + s_wait_asynccnt (ASYNCcnt path).
// ---------------------------------------------------------------------------
__global__ __launch_bounds__(256) void readout_kernel(
    const __bf16* __restrict__ Hbf, const __bf16* __restrict__ Wbf,
    const int* __restrict__ tgt, float* __restrict__ s_gold,
    float* __restrict__ sumexp) {
  __shared__ __bf16 A[16][H_];   // 16 KB, contiguous image of the A tile
  __shared__ float rowsum[16];
  __shared__ int gold[16];
  const int tid = threadIdx.x;
  const int m0 = blockIdx.x * 16;
  {
    // async-copy the contiguous 16 KB A tile: 4 x b128 per thread
    const __bf16* gsrc = Hbf + (size_t)m0 * H_;
    unsigned lbase = (unsigned)(unsigned long long)(&A[0][0]);
#pragma unroll
    for (int i = 0; i < 4; ++i) {
      unsigned off = (unsigned)(tid * 16 + i * 4096);
      unsigned lds = lbase + off;
      unsigned long long ga = (unsigned long long)gsrc + off;
      asm volatile("global_load_async_to_lds_b128 %0, %1, off"
                   :
                   : "v"(lds), "v"(ga)
                   : "memory");
    }
  }
  if (tid < 16) {
    rowsum[tid] = 0.0f;
    gold[tid] = tgt[m0 + tid + 8];
  }
  asm volatile("s_wait_asynccnt 0x0" ::: "memory");
  __syncthreads();
  const int lane = tid & 31, wave = tid >> 5;
  const int l15 = lane & 15;
  const int half8 = (lane >> 4) << 3;
  const int hi8 = (lane >> 4) << 3;
  for (int ti = 0; ti < 5; ++ti) {
    int n0 = (blockIdx.y * 40 + wave * 5 + ti) * 16;
    const __bf16* Brow = Wbf + (size_t)(n0 + l15) * H_;
    v8f acc = {};
    for (int k0 = 0; k0 < H_; k0 += 32) {
      v16bf a = ldfrag_bf(&A[l15][0], k0, half8);
      v16bf b = ldfrag_bf(Brow, k0, half8);
      acc = __builtin_amdgcn_wmma_f32_16x16x32_bf16(false, a, false, b,
                                                    (short)0, acc, false,
                                                    false);
    }
    int col = n0 + l15;
#pragma unroll
    for (int r2 = 0; r2 < 8; ++r2) {
      float v = acc[r2];
      int li = hi8 + r2;                      // local row 0..15
      if (gold[li] == col) s_gold[m0 + li] = v;
      float ev = __expf(v);
      ev += __shfl_xor(ev, 1, 32);            // reduce within 16-lane half
      ev += __shfl_xor(ev, 2, 32);
      ev += __shfl_xor(ev, 4, 32);
      ev += __shfl_xor(ev, 8, 32);
      if (l15 == 0) atomicAdd(&rowsum[li], ev);
    }
  }
  __syncthreads();
  if (tid < 16) atomicAdd(&sumexp[m0 + tid], rowsum[tid]);
}

// ---------------------------------------------------------------------------
// K6: combine everything into the two per-batch log-prob sums.
// ---------------------------------------------------------------------------
__global__ __launch_bounds__(256) void final_kernel(
    const int* __restrict__ tgt, const float* __restrict__ s_gold,
    const float* __restrict__ sumexp, const float* __restrict__ a_sent_arr,
    const float* __restrict__ pptr_gold, float* __restrict__ out) {
  int r = blockIdx.x * 256 + threadIdx.x;
  if (r >= NDEC_) return;
  int b = r & 7;
  int g = tgt[r + 8];
  if (g == 0) return;  // pad mask
  float pv = __expf(s_gold[r]) / sumexp[r];
  float as = a_sent_arr[r];
  float pp = pptr_gold[r];
  atomicAdd(&out[b], logf(pp + pv * as));
  atomicAdd(&out[8 + b], logf(pp + as));
}

// ---------------------------------------------------------------------------
extern "C" void kernel_launch(void* const* d_in, const int* in_sizes, int n_in,
                              void* d_out, int out_size, void* d_ws,
                              size_t ws_size, hipStream_t stream) {
  const int* src = (const int*)d_in[0];
  const int* tgt = (const int*)d_in[1];
  const float* emb = (const float*)d_in[2];
  const float* W_ih = (const float*)d_in[3];
  const float* W_hh = (const float*)d_in[4];
  const float* b_ih = (const float*)d_in[5];
  const float* b_hh = (const float*)d_in[6];
  const float* W_att = (const float*)d_in[7];
  const float* sent = (const float*)d_in[8];
  const float* W_read = (const float*)d_in[9];
  float* out = (float*)d_out;

  float* ws = (float*)d_ws;
  float* Xg = ws;                              // 1536*2048 f32
  float* H_all = Xg + (size_t)NROW_ * G4_;     // 1536*512  f32
  float* Q = H_all + (size_t)NROW_ * H_;       // 768*512   f32
  float* a_sent = Q + (size_t)NDEC_ * H_;      // 768
  float* pptr = a_sent + NDEC_;                // 768
  float* s_gold = pptr + NDEC_;                // 768
  float* sumexp = s_gold + NDEC_;              // 768
  __bf16* Hbf = (__bf16*)(sumexp + NDEC_);     // 768*512   bf16
  __bf16* Wbf = Hbf + (size_t)NDEC_ * H_;      // 32000*512 bf16 (~51 MB total)

  hipLaunchKernelGGL(init_kernel, dim3(3), dim3(256), 0, stream, out, sumexp);
  hipLaunchKernelGGL(wconv_kernel, dim3(4096), dim3(256), 0, stream, W_read,
                     Wbf);
  hipLaunchKernelGGL(xgates_kernel, dim3(96, 8), dim3(256), 0, stream, src, tgt,
                     emb, W_ih, Xg);
  hipLaunchKernelGGL(lstm_kernel, dim3(1), dim3(512), 0, stream, Xg, W_hh, b_ih,
                     b_hh, H_all);
  hipLaunchKernelGGL(hconv_kernel, dim3(NDEC_ * H_ / 256), dim3(256), 0, stream,
                     H_all, Hbf);
  hipLaunchKernelGGL(q_kernel, dim3(192), dim3(256), 0, stream, H_all, W_att,
                     Q);
  hipLaunchKernelGGL(att_kernel, dim3(768), dim3(256), 0, stream, H_all, Q,
                     sent, src, tgt, a_sent, pptr);
  hipLaunchKernelGGL(readout_kernel, dim3(48, 50), dim3(256), 0, stream, Hbf,
                     Wbf, tgt, s_gold, sumexp);
  hipLaunchKernelGGL(final_kernel, dim3(3), dim3(256), 0, stream, tgt, s_gold,
                     sumexp, a_sent, pptr, out);
}
